// GCNN_10110353014832
// MI455X (gfx1250) — matrix operand order repaired and checked by reading, not compiled, not run on hardware
//
#include <hip/hip_runtime.h>
#include <hip/hip_bf16.h>

typedef __attribute__((ext_vector_type(16))) _Float16 v16h;
typedef __attribute__((ext_vector_type(8)))  _Float16 v8h;
typedef __attribute__((ext_vector_type(8)))  float    v8f;
typedef __attribute__((ext_vector_type(4)))  float    f4;
typedef __attribute__((ext_vector_type(4)))  int      i4;

#define NB 1024   // batch
#define NS 256    // sites
#define NG 256    // group elements
#define NF 32     // feats

#if __has_builtin(__builtin_amdgcn_global_load_async_to_lds_b128) && \
    __has_builtin(__builtin_amdgcn_s_wait_asynccnt)
#define HAVE_ASYNC_LDS 1
#else
#define HAVE_ASYNC_LDS 0
#endif

// ---------------- conversion / packing ----------------

__global__ __launch_bounds__(256) void k_cvt_xin(const float* __restrict__ x,
                                                 _Float16* __restrict__ o) {
  int i = blockIdx.x * 256 + threadIdx.x;   // 1024*256 elems, grid=1024
  o[i] = (_Float16)x[i];
}

// Pack shifted w_symm into WMMA-B lane layout:
// wsp[(((g*8 + c)*2 + nt)*32 + lane)*16 + h] = w_symm[(c*32 + h + 8*(lane>>4) + 8*(h>>3) + g) & 255][(lane&15)+16*nt]
__global__ __launch_bounds__(256) void k_pack_wsym(const float* __restrict__ w,
                                                   _Float16* __restrict__ o) {
  int idx = blockIdx.x * 256 + threadIdx.x;          // 2,097,152 halves, grid=8192
  int h = idx & 15, lane = (idx >> 4) & 31, nt = (idx >> 9) & 1;
  int c = (idx >> 10) & 7, g = (idx >> 13) & 255;
  int row = (c * 32 + h + 8 * (lane >> 4) + 8 * (h >> 3) + g) & 255;
  int col = (lane & 15) + 16 * nt;
  o[idx] = (_Float16)w[row * 32 + col];
}

// Pack w_eq into WMMA-B lane layout per (layer, g):
// weqp[(((l*256+g)*2 + nt)*32 + lane)*16 + h] = w_eq[l][g][h + 8*(lane>>4) + 8*(h>>3)][(lane&15)+16*nt]
__global__ __launch_bounds__(256) void k_pack_weq(const float* __restrict__ w,
                                                  _Float16* __restrict__ o) {
  int idx = blockIdx.x * 256 + threadIdx.x;          // 786,432 halves, grid=3072
  int h = idx & 15, lane = (idx >> 4) & 31, nt = (idx >> 9) & 1;
  int g = (idx >> 10) & 255, l = idx >> 18;
  int fi = h + 8 * (lane >> 4) + 8 * (h >> 3);
  int fo = (lane & 15) + 16 * nt;
  o[idx] = (_Float16)w[(((size_t)l * 256 + g) * 32 + fi) * 32 + fo];
}

// ---------------- DenseSymm: x0[b][g][f] = sum_i xin[b][i]*w_symm[(i+g)%256][f] + b_symm[f]
__global__ __launch_bounds__(256) void k_densesymm(const _Float16* __restrict__ xin, // [1024][256]
                                                   const _Float16* __restrict__ wsp,
                                                   const float* __restrict__ bsym,   // [32]
                                                   _Float16* __restrict__ xout) {    // [1024][256][32]
  __shared__ __align__(16) _Float16 xs[32 * 264];    // 32 b rows, stride 264 halves (bank-safe)
  const int t = threadIdx.x, wave = t >> 5, lane = t & 31;
  const int b0 = blockIdx.x * 32;
  const int go = blockIdx.y * 8 + wave;
  const int m = lane & 15, kg = lane >> 4, ak0 = kg * 8;

  // stage 32 xin rows (256 halves each) into LDS
  #pragma unroll
  for (int v = 0; v < 4; ++v) {
    int u = v * 256 + t;            // 0..1023 16B units
    int bl = u >> 5, off = u & 31;
    *(f4*)(xs + bl * 264 + off * 8) = *(const f4*)(xin + (size_t)(b0 + bl) * 256 + off * 8);
  }
  __syncthreads();

  v8f acc[2][2] = {};
  for (int c = 0; c < 8; ++c) {
    const _Float16* wp = wsp + (size_t)(go * 8 + c) * 1024;
    v16h bf0 = *(const v16h*)(wp + lane * 16);
    v16h bf1 = *(const v16h*)(wp + 512 + lane * 16);
    #pragma unroll
    for (int mt = 0; mt < 2; ++mt) {
      const _Float16* ap = xs + (mt * 16 + m) * 264 + c * 32;
      v8h lo = *(const v8h*)(ap + ak0);
      v8h hi = *(const v8h*)(ap + ak0 + 16);
      v16h a;
      #pragma unroll
      for (int q = 0; q < 8; ++q) { a[q] = lo[q]; a[q + 8] = hi[q]; }
      acc[mt][0] = __builtin_amdgcn_wmma_f32_16x16x32_f16(false, a, false, bf0, (short)0, acc[mt][0], false, false);
      acc[mt][1] = __builtin_amdgcn_wmma_f32_16x16x32_f16(false, a, false, bf1, (short)0, acc[mt][1], false, false);
    }
  }

  #pragma unroll
  for (int mt = 0; mt < 2; ++mt)
    #pragma unroll
    for (int nt = 0; nt < 2; ++nt) {
      int fo = (lane & 15) + 16 * nt;
      float bv = bsym[fo];
      #pragma unroll
      for (int j = 0; j < 8; ++j) {
        int b = b0 + mt * 16 + j + 8 * kg;       // C-matrix layout: M = j + 8*(lane>>4)
        float v = acc[mt][nt][j] + bv;
        xout[((size_t)b * NG + go) * NF + fo] = (_Float16)v;
      }
    }
}

// ---------------- Equivariant layer: group convolution + bias + swish ----------------
// Double-buffered LDS chunks of 8 gi, streamed via async-to-LDS when available.

__device__ __forceinline__ void eq_stage_chunk(const _Float16* __restrict__ xin,
                                               _Float16* buf, int b0, int c, int t) {
  #pragma unroll
  for (int v = 0; v < 4; ++v) {
    int u = v * 256 + t;                 // 0..1023 16B units (32b x 8gi x 32fi)
    int bl = u >> 5, off = u & 31;       // 32 units per b row (256 halves)
    const _Float16* g = xin + (size_t)(b0 + bl) * (NG * NF) + c * 256 + off * 8;
    _Float16* l = buf + bl * 264 + off * 8;
#if HAVE_ASYNC_LDS
    __builtin_amdgcn_global_load_async_to_lds_b128(
        (__attribute__((address_space(1))) i4*)g,
        (__attribute__((address_space(3))) i4*)l, 0, 0);
#else
    *(f4*)l = *(const f4*)g;
#endif
  }
}

__device__ __forceinline__ void eq_compute_chunk(const _Float16* bufp,
                                                 const _Float16* __restrict__ wp_l,
                                                 int c, int go, int m, int ak0, int lane,
                                                 v8f acc[2][2]) {
  for (int gil = 0; gil < 8; ++gil) {
    int gi = c * 8 + gil;
    int g = (go - gi) & 255;                       // group_algebra[gi, go]
    const _Float16* wp = wp_l + (size_t)g * 1024;
    v16h bf0 = *(const v16h*)(wp + lane * 16);
    v16h bf1 = *(const v16h*)(wp + 512 + lane * 16);
    #pragma unroll
    for (int mt = 0; mt < 2; ++mt) {
      const _Float16* ap = bufp + (mt * 16 + m) * 264 + gil * 32;
      v8h lo = *(const v8h*)(ap + ak0);
      v8h hi = *(const v8h*)(ap + ak0 + 16);
      v16h a;
      #pragma unroll
      for (int q = 0; q < 8; ++q) { a[q] = lo[q]; a[q + 8] = hi[q]; }
      acc[mt][0] = __builtin_amdgcn_wmma_f32_16x16x32_f16(false, a, false, bf0, (short)0, acc[mt][0], false, false);
      acc[mt][1] = __builtin_amdgcn_wmma_f32_16x16x32_f16(false, a, false, bf1, (short)0, acc[mt][1], false, false);
    }
  }
}

__global__ __launch_bounds__(256) void k_eq_layer(const _Float16* __restrict__ xin,  // [1024][256][32]
                                                  const _Float16* __restrict__ wp_l, // packed [256][2][32][16]
                                                  const float* __restrict__ beq,     // [32]
                                                  _Float16* __restrict__ xout) {     // [1024][256][32]
  __shared__ __align__(16) _Float16 xs[2 * 32 * 264]; // two 16.5KB chunk buffers
  const int t = threadIdx.x, wave = t >> 5, lane = t & 31;
  const int b0 = blockIdx.x * 32;
  const int go = blockIdx.y * 8 + wave;
  const int m = lane & 15, kg = lane >> 4, ak0 = kg * 8;

  v8f acc[2][2] = {};

  eq_stage_chunk(xin, xs, b0, 0, t);               // prologue: chunk 0 -> buf 0
  for (int c = 0; c < 31; ++c) {
    const int cur = c & 1;
    if (c > 0) __syncthreads();                    // all waves done reading buf[1-cur]
    eq_stage_chunk(xin, xs + (1 - cur) * 8448, b0, c + 1, t);
#if HAVE_ASYNC_LDS
    __builtin_amdgcn_s_wait_asynccnt(4);           // retire chunk c, keep chunk c+1 in flight
#endif
    __syncthreads();                               // chunk c visible to all waves
    eq_compute_chunk(xs + cur * 8448, wp_l, c, go, m, ak0, lane, acc);
  }
#if HAVE_ASYNC_LDS
  __builtin_amdgcn_s_wait_asynccnt(0);
#endif
  __syncthreads();
  eq_compute_chunk(xs + 8448, wp_l, 31, go, m, ak0, lane, acc);   // 31 & 1 == 1

  #pragma unroll
  for (int mt = 0; mt < 2; ++mt)
    #pragma unroll
    for (int nt = 0; nt < 2; ++nt) {
      int fo = (lane & 15) + 16 * nt;
      float bv = beq[fo];
      #pragma unroll
      for (int j = 0; j < 8; ++j) {
        int b = b0 + mt * 16 + j + 8 * kg;
        float v = acc[mt][nt][j] + bv;
        v = v * (1.0f / (1.0f + __expf(-v)));      // swish
        xout[((size_t)b * NG + go) * NF + fo] = (_Float16)v;
      }
    }
}

// ---------------- final mean over (feats * n_symm) ----------------
__global__ __launch_bounds__(256) void k_reduce(const _Float16* __restrict__ x,
                                                float* __restrict__ out) {
  __shared__ float red[256];
  int b = blockIdx.x, t = threadIdx.x;
  const _Float16* p = x + (size_t)b * (NG * NF) + t * 32;
  float s = 0.f;
  #pragma unroll
  for (int v = 0; v < 4; ++v) {
    v8h h = *(const v8h*)(p + v * 8);
    #pragma unroll
    for (int q = 0; q < 8; ++q) s += (float)h[q];
  }
  red[t] = s;
  __syncthreads();
  for (int st = 128; st > 0; st >>= 1) {
    if (t < st) red[t] += red[t + st];
    __syncthreads();
  }
  if (t == 0) out[b] = red[0] * (1.0f / 8192.0f);
}

// ---------------- host ----------------
extern "C" void kernel_launch(void* const* d_in, const int* in_sizes, int n_in,
                              void* d_out, int out_size, void* d_ws, size_t ws_size,
                              hipStream_t stream) {
  (void)in_sizes; (void)n_in; (void)out_size; (void)ws_size;
  const float* x_in   = (const float*)d_in[0];
  // d_in[1] = perms (cyclic shifts), d_in[2] = group_algebra ((go-gi)%256): structure used analytically
  const float* w_symm = (const float*)d_in[3];
  const float* b_symm = (const float*)d_in[4];
  const float* w_eq   = (const float*)d_in[5];
  const float* b_eq   = (const float*)d_in[6];
  float* out = (float*)d_out;

  char* ws = (char*)d_ws;
  _Float16* xin_h = (_Float16*)(ws + 0);          // 512 KB
  _Float16* wsp   = (_Float16*)(ws + 524288);     // 4 MB   packed shifted w_symm
  _Float16* weqp  = (_Float16*)(ws + 4718592);    // 1.5 MB packed w_eq (3 layers)
  _Float16* xb0   = (_Float16*)(ws + 6291456);    // 16 MB  activations ping
  _Float16* xb1   = (_Float16*)(ws + 23068672);   // 16 MB  activations pong

  hipLaunchKernelGGL(k_cvt_xin,   dim3(1024),   dim3(256), 0, stream, x_in, xin_h);
  hipLaunchKernelGGL(k_pack_wsym, dim3(8192),   dim3(256), 0, stream, w_symm, wsp);
  hipLaunchKernelGGL(k_pack_weq,  dim3(3072),   dim3(256), 0, stream, w_eq, weqp);

  hipLaunchKernelGGL(k_densesymm, dim3(32, 32), dim3(256), 0, stream, xin_h, wsp, b_symm, xb0);

  hipLaunchKernelGGL(k_eq_layer,  dim3(32, 32), dim3(256), 0, stream, xb0, weqp,          b_eq,      xb1);
  hipLaunchKernelGGL(k_eq_layer,  dim3(32, 32), dim3(256), 0, stream, xb1, weqp + 262144, b_eq + 32, xb0);
  hipLaunchKernelGGL(k_eq_layer,  dim3(32, 32), dim3(256), 0, stream, xb0, weqp + 524288, b_eq + 64, xb1);

  hipLaunchKernelGGL(k_reduce,    dim3(1024),   dim3(256), 0, stream, xb1, out);
}
